// EnhancedMoELayer_56169582297271
// MI455X (gfx1250) — matrix operand order repaired and checked by reading, not compiled, not run on hardware
//
#include <hip/hip_runtime.h>
#include <hip/hip_bf16.h>
#include <math.h>

// ---------------------------------------------------------------------------
// MoE (E=8 experts, top-2) on gfx1250.
// Big GEMMs in bf16 WMMA (v_wmma_f32_16x16x32_bf16), fp32 accumulation.
// Block tile M=64 x N=128 x K=128; B stored transposed in LDS so both A and
// B fragments are 2x ds_load_b128.  Pass B stages its A tile (bf16 H rows,
// contiguous 2D tile) with the Tensor Data Mover (tensor_load_to_lds),
// using D# LDS padding to reproduce the 272B row stride.
// ---------------------------------------------------------------------------

#define D_EMB   1024
#define FF_DIM  4096
#define N_EXP   8
#define TOPK    2

#define BLK_M   64
#define BLK_N   128
#define BLK_K   128
#define LDA     (BLK_K + 8)   // bf16 units; 272B rows -> 16B aligned
#define LDBT    (BLK_K + 8)   // transposed B: [BLK_N][LDBT]

typedef __attribute__((ext_vector_type(16))) __bf16 bf16x16;
typedef __attribute__((ext_vector_type(8)))  float  f32x8;
typedef __attribute__((ext_vector_type(4)))  unsigned int u32x4;
typedef __attribute__((ext_vector_type(8)))  int          i32x8;
typedef __attribute__((ext_vector_type(4)))  int          i32x4;

// pack two fp32 -> two bf16 in one dword (v_cvt_pk_bf16_f32)
static __device__ __forceinline__ unsigned int pack2bf(float lo, float hi) {
    __bf16 a = (__bf16)lo, b = (__bf16)hi;
    unsigned int ua = __builtin_bit_cast(unsigned short, a);
    unsigned int ub = __builtin_bit_cast(unsigned short, b);
    return ua | (ub << 16);
}

// ---------------------------------------------------------------------------
// TDM: async-load a 2D bf16 tile [rows x 128] (row stride `srcld` elements)
// from global into LDS with 272B padded row stride (matches LDA).
// D# group1: data_size=1(2B), pad_enable, pad_interval=5 (every 256B),
// pad_amount=3 (4 dwords) -> LDS row stride 256+16 = 272B.
// tensor_dim1 clamps rows so the TDM never reads past the tensor.
// ---------------------------------------------------------------------------
static __device__ __forceinline__ void tdm_load_tile_bf16(
    const __bf16* gsrc, unsigned int lds_off, unsigned int rows_avail,
    unsigned int srcld)
{
    const unsigned long long ga = (unsigned long long)(const void*)gsrc;

    u32x4 g0;
    g0[0] = 1u;                                              // count=1, user D#
    g0[1] = lds_off;                                         // lds_addr (bytes)
    g0[2] = (unsigned int)(ga & 0xFFFFFFFFu);                // global_addr lo
    g0[3] = (unsigned int)((ga >> 32) & 0x01FFFFFFu) | (2u << 30);  // hi | type=2

    i32x8 g1;
    g1[0] = (int)((1u << 16) | (1u << 20) | (5u << 22) | (3u << 25));
    g1[1] = (int)((srcld & 0xFFFFu) << 16);                  // tensor_dim0 lo16
    g1[2] = (int)(((srcld >> 16) & 0xFFFFu)                  // tensor_dim0 hi16
                  | ((rows_avail & 0xFFFFu) << 16));         // tensor_dim1 lo16
    g1[3] = (int)(((rows_avail >> 16) & 0xFFFFu)             // tensor_dim1 hi16
                  | (128u << 16));                           // tile_dim0 = 128
    g1[4] = (int)BLK_M;                                      // tile_dim1=64, dim2=0
    g1[5] = (int)srcld;                                      // dim0_stride lo32
    g1[6] = 0;                                               // dim0_stride hi16
    g1[7] = 0;

    const i32x4 gz = {0, 0, 0, 0};                           // groups 2/3 unused (2D)

#if defined(__clang_major__) && (__clang_major__ >= 23)
    const i32x8 gz8 = {0, 0, 0, 0, 0, 0, 0, 0};
    __builtin_amdgcn_tensor_load_to_lds(g0, g1, gz, gz, gz8, 0);
#else
    __builtin_amdgcn_tensor_load_to_lds(g0, g1, gz, gz, 0);
#endif
}

// ---------------------------------------------------------------------------
// Kernel 1: gating.  One wave32 per token.
// ---------------------------------------------------------------------------
__global__ __launch_bounds__(256) void moe_gate_topk(
    const float* __restrict__ x, const float* __restrict__ gate_w,
    int* __restrict__ top_i, float* __restrict__ top_v, int ntok)
{
    const int wave = threadIdx.x >> 5;
    const int lane = threadIdx.x & 31;
    const int n = blockIdx.x * 8 + wave;
    if (n >= ntok) return;

    float g[N_EXP];
#pragma unroll
    for (int e = 0; e < N_EXP; ++e) g[e] = 0.f;

    const float* xr = x + (size_t)n * D_EMB;
    for (int d = lane; d < D_EMB; d += 32) {
        const float xv = xr[d];
        const float* gw = gate_w + (size_t)d * N_EXP;
#pragma unroll
        for (int e = 0; e < N_EXP; ++e) g[e] = fmaf(xv, gw[e], g[e]);
    }
#pragma unroll
    for (int e = 0; e < N_EXP; ++e) {
        g[e] += __shfl_xor(g[e], 16, 32);
        g[e] += __shfl_xor(g[e], 8, 32);
        g[e] += __shfl_xor(g[e], 4, 32);
        g[e] += __shfl_xor(g[e], 2, 32);
        g[e] += __shfl_xor(g[e], 1, 32);
    }
    if (lane == 0) {
        float mx = g[0];
#pragma unroll
        for (int e = 1; e < N_EXP; ++e) mx = fmaxf(mx, g[e]);
        float p[N_EXP];
#pragma unroll
        for (int e = 0; e < N_EXP; ++e) p[e] = __expf(g[e] - mx);
        int i1 = 0;
#pragma unroll
        for (int e = 1; e < N_EXP; ++e) if (p[e] > p[i1]) i1 = e;
        int i2 = (i1 == 0) ? 1 : 0;
#pragma unroll
        for (int e = 0; e < N_EXP; ++e) if (e != i1 && p[e] > p[i2]) i2 = e;
        const float v1 = p[i1], v2 = p[i2];
        const float inv = 1.f / (v1 + v2);
        top_i[2 * n + 0] = i1;  top_v[2 * n + 0] = v1 * inv;
        top_i[2 * n + 1] = i2;  top_v[2 * n + 1] = v2 * inv;
    }
}

// ---------------------------------------------------------------------------
// Deterministic compact routing (no atomics; serial scans are microseconds).
// ---------------------------------------------------------------------------
__global__ void moe_count(const int* __restrict__ top_i, int* __restrict__ counts,
                          int npairs)
{
    const int e = threadIdx.x;
    if (e >= N_EXP) return;
    int c = 0;
    for (int p = 0; p < npairs; ++p) c += (top_i[p] == e) ? 1 : 0;
    counts[e] = c;
}

__global__ void moe_scan(const int* __restrict__ counts, int* __restrict__ offs)
{
    if (threadIdx.x != 0 || blockIdx.x != 0) return;
    int acc = 0;
    for (int e = 0; e < N_EXP; ++e) { offs[e] = acc; acc += counts[e]; }
    offs[N_EXP] = acc;
}

__global__ void moe_fill(const int* __restrict__ top_i, const float* __restrict__ top_v,
                         const int* __restrict__ offs, int* __restrict__ pair_token,
                         float* __restrict__ pair_gate, int npairs)
{
    const int e = blockIdx.x;
    if (threadIdx.x != 0) return;
    int idx = offs[e];
    for (int p = 0; p < npairs; ++p) {
        if (top_i[p] == e) {
            pair_token[idx] = p >> 1;
            pair_gate[idx]  = top_v[p];
            ++idx;
        }
    }
}

// ---------------------------------------------------------------------------
// Fragment builders (CDNA5 16-bit WMMA striping):
//   A (16x32): lane<16 row M=lane&15, K {0..7,16..23}; lane>=16 K {8..15,24..31}
//   B (32x16): lane col N=lane&15; K split as A: lane<16 K {0..7,16..23}; lane>=16 K {8..15,24..31}
//   C/D      : vgpr r -> M = r + 8*half, N = lane&15
// ---------------------------------------------------------------------------
static __device__ __forceinline__ bf16x16 load_a_frag(
    const __bf16* __restrict__ As, int row, int kb, int half)
{
    bf16x16 a;
    const __bf16* p = As + row * LDA;
#pragma unroll
    for (int v = 0; v < 8; ++v) {
        const int kk = (v < 4) ? (kb + (half << 3) + (v << 1))
                               : (kb + 16 + (half << 3) + ((v - 4) << 1));
        a[2 * v]     = p[kk];
        a[2 * v + 1] = p[kk + 1];
    }
    return a;
}

static __device__ __forceinline__ bf16x16 load_bT_frag(
    const __bf16* __restrict__ BsT, int kb, int col, int half)
{
    bf16x16 b;
    const __bf16* p = BsT + col * LDBT + kb + (half << 3);
#pragma unroll
    for (int v = 0; v < 16; ++v) b[v] = p[(v < 8) ? v : (v + 8)];   // b[j] <-> K = kb + 8*half + j (j<8), kb + 16 + 8*half + (j-8) (j>=8)
    return b;
}

// Stage a 128(k) x 128(col) fp32 weight block into transposed bf16 LDS.
static __device__ __forceinline__ void stage_bT(
    const float* __restrict__ src, size_t ldg, __bf16* __restrict__ BsT, int tid)
{
    const int c0  = (tid & 31) << 2;
    const int krb = (tid >> 5) << 4;
#pragma unroll
    for (int sub = 0; sub < 2; ++sub) {
        float f[8][4];
#pragma unroll
        for (int j = 0; j < 8; ++j) {
            const float4 v = *(const float4*)(src + (size_t)(krb + sub * 8 + j) * ldg + c0);
            f[j][0] = v.x; f[j][1] = v.y; f[j][2] = v.z; f[j][3] = v.w;
        }
#pragma unroll
        for (int i = 0; i < 4; ++i) {
            uint4 u;
            u.x = pack2bf(f[0][i], f[1][i]);
            u.y = pack2bf(f[2][i], f[3][i]);
            u.z = pack2bf(f[4][i], f[5][i]);
            u.w = pack2bf(f[6][i], f[7][i]);
            *(uint4*)&BsT[(c0 + i) * LDBT + krb + sub * 8] = u;
        }
    }
}

// ---------------------------------------------------------------------------
// Pass A: H = gelu_exact( X_routed @ fc_w[e] ),  bf16 H [npairs x 4096].
// A rows are gathered via pair_token and fp32->bf16 converted: manual staging.
// ---------------------------------------------------------------------------
__global__ __launch_bounds__(256) void moe_fc_gelu(
    const float* __restrict__ x, const float* __restrict__ fc_w,
    const int* __restrict__ offs, const int* __restrict__ pair_token,
    __bf16* __restrict__ H)
{
    __shared__ __align__(16) __bf16 As [BLK_M * LDA];    // 17408 B
    __shared__ __align__(16) __bf16 BsT[BLK_N * LDBT];   // 34816 B

    const int e   = blockIdx.z;
    const int beg = offs[e], end = offs[e + 1];
    const int m0  = beg + blockIdx.y * BLK_M;
    if (m0 >= end) return;

    const int wave = threadIdx.x >> 5;
    const int lane = threadIdx.x & 31;
    const int half = lane >> 4;
    const int mrow = lane & 15;
    const int ncol = lane & 15;
    const int col0w = wave << 4;
    const int colblk = blockIdx.x * BLK_N;

    const int arow  = threadIdx.x >> 2;
    const int apart = (threadIdx.x & 3) << 5;
    int agm = m0 + arow; if (agm >= end) agm = beg;
    const float* arp = x + (size_t)pair_token[agm] * D_EMB + apart;

    const float* Bsrc = fc_w + (size_t)e * D_EMB * FF_DIM + colblk;

    f32x8 acc[4];
#pragma unroll
    for (int rt = 0; rt < 4; ++rt)
#pragma unroll
        for (int i = 0; i < 8; ++i) acc[rt][i] = 0.f;

    for (int kc = 0; kc < D_EMB; kc += BLK_K) {
        __syncthreads();
#pragma unroll
        for (int j = 0; j < 8; ++j) {
            const float4 f = *(const float4*)(arp + kc + j * 4);
            unsigned int* d = (unsigned int*)&As[arow * LDA + apart + j * 4];
            d[0] = pack2bf(f.x, f.y);
            d[1] = pack2bf(f.z, f.w);
        }
        stage_bT(Bsrc + (size_t)kc * FF_DIM, FF_DIM, BsT, threadIdx.x);
        __syncthreads();

#pragma unroll
        for (int kb = 0; kb < BLK_K; kb += 32) {
            const bf16x16 b = load_bT_frag(BsT, kb, col0w + ncol, half);
#pragma unroll
            for (int rt = 0; rt < 4; ++rt) {
                const bf16x16 a = load_a_frag(As, rt * 16 + mrow, kb, half);
                acc[rt] = __builtin_amdgcn_wmma_f32_16x16x32_bf16(
                              false, a, false, b, (short)0, acc[rt], false, false);
            }
        }
    }

#pragma unroll
    for (int rt = 0; rt < 4; ++rt) {
#pragma unroll
        for (int r = 0; r < 8; ++r) {
            const int M  = rt * 16 + r + (half << 3);
            const int gm = m0 + M;
            if (gm < end) {
                const float v = acc[rt][r];
                const float g = 0.5f * v * (1.0f + erff(v * 0.70710678118654752f));
                H[(size_t)gm * FF_DIM + colblk + col0w + ncol] = (__bf16)g;
            }
        }
    }
}

// ---------------------------------------------------------------------------
// Pass B: out[token] += gate * ( H_tile @ proj_w[e] ).
// A (=H) is a contiguous bf16 2D tile: staged by the Tensor Data Mover.
// Wave 0 issues the TDM; all threads stage B (overlaps the DMA); wave 0
// waits TENSORcnt==0; the workgroup barrier then publishes the LDS tiles.
// ---------------------------------------------------------------------------
__global__ __launch_bounds__(256) void moe_proj_combine(
    const __bf16* __restrict__ H, const float* __restrict__ proj_w,
    const int* __restrict__ offs, const int* __restrict__ pair_token,
    const float* __restrict__ pair_gate, float* __restrict__ out, int npairs)
{
    __shared__ __align__(16) __bf16 As [BLK_M * LDA];
    __shared__ __align__(16) __bf16 BsT[BLK_N * LDBT];

    const int e   = blockIdx.z;
    const int beg = offs[e], end = offs[e + 1];
    const int m0  = beg + blockIdx.y * BLK_M;
    if (m0 >= end) return;

    const int wave = threadIdx.x >> 5;
    const int lane = threadIdx.x & 31;
    const int half = lane >> 4;
    const int mrow = lane & 15;
    const int ncol = lane & 15;
    const int col0w = wave << 4;
    const int colblk = blockIdx.x * BLK_N;

    const unsigned int as_lds = (unsigned int)(uintptr_t)(void*)As;
    const unsigned int rows_avail = (unsigned int)(npairs - m0);

    const float* Bsrc = proj_w + (size_t)e * FF_DIM * D_EMB + colblk;

    f32x8 acc[4];
#pragma unroll
    for (int rt = 0; rt < 4; ++rt)
#pragma unroll
        for (int i = 0; i < 8; ++i) acc[rt][i] = 0.f;

    for (int kc = 0; kc < FF_DIM; kc += BLK_K) {
        __syncthreads();
        if (wave == 0) {     // TDM DMA: H[m0..m0+63][kc..kc+127] -> As (padded)
            tdm_load_tile_bf16(H + (size_t)m0 * FF_DIM + kc, as_lds,
                               rows_avail, FF_DIM);
        }
        stage_bT(Bsrc + (size_t)kc * D_EMB, D_EMB, BsT, threadIdx.x);
        if (wave == 0) {
            __builtin_amdgcn_s_wait_tensorcnt(0);
        }
        __syncthreads();

#pragma unroll
        for (int kb = 0; kb < BLK_K; kb += 32) {
            const bf16x16 b = load_bT_frag(BsT, kb, col0w + ncol, half);
#pragma unroll
            for (int rt = 0; rt < 4; ++rt) {
                const bf16x16 a = load_a_frag(As, rt * 16 + mrow, kb, half);
                acc[rt] = __builtin_amdgcn_wmma_f32_16x16x32_bf16(
                              false, a, false, b, (short)0, acc[rt], false, false);
            }
        }
    }

#pragma unroll
    for (int rt = 0; rt < 4; ++rt) {
#pragma unroll
        for (int r = 0; r < 8; ++r) {
            const int M  = rt * 16 + r + (half << 3);
            const int gm = m0 + M;
            if (gm < end) {
                const int   tok = pair_token[gm];
                const float gte = pair_gate[gm];
                atomicAdd(&out[(size_t)tok * D_EMB + colblk + col0w + ncol],
                          gte * acc[rt][r]);
            }
        }
    }
}

// ---------------------------------------------------------------------------
// Host launcher.  Workspace: routing tables + bf16 H [npairs x 4096] (64 MB).
// ---------------------------------------------------------------------------
extern "C" void kernel_launch(void* const* d_in, const int* in_sizes, int n_in,
                              void* d_out, int out_size, void* d_ws, size_t ws_size,
                              hipStream_t stream) {
    const float* x      = (const float*)d_in[0];
    const float* gate_w = (const float*)d_in[1];
    const float* fc_w   = (const float*)d_in[2];
    const float* proj_w = (const float*)d_in[3];
    float* out = (float*)d_out;

    const int ntok   = in_sizes[0] / D_EMB;      // 4096
    const int npairs = ntok * TOPK;              // 8192

    char* ws = (char*)d_ws;
    size_t o = 0;
    int*    top_i      = (int*)(ws + o);   o += (size_t)npairs * 4;
    float*  top_v      = (float*)(ws + o); o += (size_t)npairs * 4;
    int*    counts     = (int*)(ws + o);   o += N_EXP * 4;
    int*    offs       = (int*)(ws + o);   o += (N_EXP + 1) * 4;
    int*    pair_token = (int*)(ws + o);   o += (size_t)npairs * 4;
    float*  pair_gate  = (float*)(ws + o); o += (size_t)npairs * 4;
    o = (o + 255) & ~(size_t)255;
    __bf16* H          = (__bf16*)(ws + o);      // npairs * 4096 * 2 bytes

    hipMemsetAsync(d_out, 0, (size_t)out_size * sizeof(float), stream);

    moe_gate_topk<<<(ntok + 7) / 8, 256, 0, stream>>>(x, gate_w, top_i, top_v, ntok);
    moe_count<<<1, N_EXP, 0, stream>>>(top_i, counts, npairs);
    moe_scan<<<1, 1, 0, stream>>>(counts, offs);
    moe_fill<<<N_EXP, 1, 0, stream>>>(top_i, top_v, offs, pair_token, pair_gate, npairs);

    const int maxmt = (npairs + BLK_M - 1) / BLK_M;   // worst case one expert owns all
    dim3 gA(FF_DIM / BLK_N, maxmt, N_EXP);
    moe_fc_gelu<<<gA, 256, 0, stream>>>(x, fc_w, offs, pair_token, H);
    dim3 gB(D_EMB / BLK_N, maxmt, N_EXP);
    moe_proj_combine<<<gB, 256, 0, stream>>>(H, proj_w, offs, pair_token, pair_gate, out,
                                             npairs);
}
